// Cross_LocalAttention_18786186953467
// MI455X (gfx1250) — matrix operand order repaired and checked by reading, not compiled, not run on hardware
//
#include <hip/hip_runtime.h>
#include <math.h>

// ---- problem constants (match reference) ----
constexpr int BATCH = 8;
constexpr int HG    = 128;          // grid H
constexpr int WG    = 128;          // grid W
constexpr int DIM   = 256;
constexpr int HEADS = 8;
constexpr int HD    = 32;           // head dim
constexpr int WS    = 64;           // tokens per 8x8 window
constexpr int LTOK  = HG * WG;      // 16384 tokens per batch image
constexpr int NWIN  = BATCH * (HG / 8) * (WG / 8);   // 2048
constexpr float SCALE = 0.17677669529663687f;        // 32^-0.5

typedef __bf16 bf16;
typedef bf16  v16bf __attribute__((ext_vector_type(16)));
typedef bf16  v8bf  __attribute__((ext_vector_type(8)));
typedef float v8f   __attribute__((ext_vector_type(8)));

// One block per window (256 threads = 8 wave32). Wave w handles head w.
__global__ __launch_bounds__(256)
void win_attn_lepe_kernel(const float* __restrict__ qkv,
                          const float* __restrict__ lepe_w,
                          const float* __restrict__ lepe_b,
                          float* __restrict__ out_x,
                          float* __restrict__ out_q)
{
    // V transposed to [ch][token] in bf16 (contiguous B-fragments for P*V)
    __shared__ bf16 sVt[DIM * WS];                 // 32768 B
    // Per-wave softmax strip: 16 rows x (64+8 pad) bf16
    __shared__ bf16 sP[HEADS][16 * 72];            // 18432 B

    const int tid  = threadIdx.x;
    const int lane = tid & 31;
    const int wave = tid >> 5;                     // == head index
    const int hw   = lane >> 4;                    // half-wave select (0/1)
    const int nl   = lane & 15;                    // lane within half

    const int win = blockIdx.x;
    const int b   = win >> 8;                      // 256 windows per batch
    const int wi  = (win >> 4) & 15;
    const int wj  = win & 15;
    const int baseSp = wi * 8 * WG + wj * 8;       // top-left token row index

    const float* Qg = qkv + (size_t)(0 * BATCH + b) * LTOK * DIM;
    const float* Kg = qkv + (size_t)(1 * BATCH + b) * LTOK * DIM;
    const float* Vg = qkv + (size_t)(2 * BATCH + b) * LTOK * DIM;

    // ---------- Stage V (bf16, transposed) and emit q output ----------
    for (int idx = tid; idx < WS * DIM; idx += 256) {
        int s  = idx >> 8;                         // token in window
        int ch = idx & 255;
        int Lrow = baseSp + (s >> 3) * WG + (s & 7);
        float v = Vg[(size_t)Lrow * DIM + ch];
        sVt[ch * WS + s] = (bf16)v;
        // second output: windowed, scaled q  [win][head][s][d]
        float qv = Qg[(size_t)Lrow * DIM + ch] * SCALE;
        int h = ch >> 5, d = ch & 31;
        out_q[(((size_t)win * HEADS + h) * WS + s) * HD + d] = qv;
    }
    __syncthreads();

    const int chB = wave * HD;                     // this head's channel base

    // ---------- K B-fragments (4 column tiles), fp32 global -> bf16 ----------
    // B fragment, contiguous k order matching A: lane holds column n = lane&15, k rows 16*hw..+15.
    v16bf kfrag[4];
    #pragma unroll
    for (int tc = 0; tc < 4; ++tc) {
        int tok  = tc * 16 + nl;
        int Lrow = baseSp + (tok >> 3) * WG + (tok & 7);
        const float* p = Kg + (size_t)Lrow * DIM + chB + hw * 16;
        #pragma unroll
        for (int i = 0; i < 16; ++i) kfrag[tc][i] = (bf16)p[i];
    }

    bf16* Pw = sP[wave];

    #pragma unroll 1
    for (int tr = 0; tr < 4; ++tr) {
        // ---- A fragment of Q (scaled): lane holds row m = lane&15 ----
        int qtok  = tr * 16 + nl;
        int qLrow = baseSp + (qtok >> 3) * WG + (qtok & 7);
        const float* qp = Qg + (size_t)qLrow * DIM + chB;
        v16bf afrag;
        #pragma unroll
        for (int i = 0; i < 8; ++i) {
            afrag[i]     = (bf16)(qp[hw * 16 + i]     * SCALE);
            afrag[i + 8] = (bf16)(qp[hw * 16 + 8 + i] * SCALE);
        }

        // ---- S strip: 4 tiles of 16x16, K=32 ----
        v8f S[4];
        #pragma unroll
        for (int tc = 0; tc < 4; ++tc) {
            v8f z = {};
            S[tc] = __builtin_amdgcn_wmma_f32_16x16x32_bf16(
                        false, afrag, false, kfrag[tc], (short)0, z, false, false);
        }

        // ---- row softmax (rows m = v + 8*hw; cols across 16 lanes + 4 tiles) ----
        #pragma unroll
        for (int v = 0; v < 8; ++v) {
            float m = fmaxf(fmaxf(S[0][v], S[1][v]), fmaxf(S[2][v], S[3][v]));
            #pragma unroll
            for (int mask = 1; mask < 16; mask <<= 1)
                m = fmaxf(m, __shfl_xor(m, mask, 32));
            float p0 = __expf(S[0][v] - m);
            float p1 = __expf(S[1][v] - m);
            float p2 = __expf(S[2][v] - m);
            float p3 = __expf(S[3][v] - m);
            float sum = p0 + p1 + p2 + p3;
            #pragma unroll
            for (int mask = 1; mask < 16; mask <<= 1)
                sum += __shfl_xor(sum, mask, 32);
            float r = __builtin_amdgcn_rcpf(sum);
            int row = v + 8 * hw;
            bf16* dst = Pw + row * 72 + nl;
            dst[0]  = (bf16)(p0 * r);
            dst[16] = (bf16)(p1 * r);
            dst[32] = (bf16)(p2 * r);
            dst[48] = (bf16)(p3 * r);
        }
        // same-wave LDS store -> cross-lane load ordering
        asm volatile("s_wait_dscnt 0" ::: "memory");

        // ---- O strip: O[16 x 32] = P[16 x 64] * V[64 x 32], fused LePE epilogue ----
        #pragma unroll
        for (int td = 0; td < 2; ++td) {
            v8f acc = {};
            #pragma unroll
            for (int kt = 0; kt < 2; ++kt) {
                // A = P rows (lane&15), k chunk: elems 0-15 at k=16*hw..16*hw+15
                const bf16* pa = Pw + nl * 72 + kt * 32 + hw * 16;
                v8bf lo = *(const v8bf*)pa;
                v8bf hi = *(const v8bf*)(pa + 8);
                v16bf ap;
                #pragma unroll
                for (int i = 0; i < 8; ++i) { ap[i] = lo[i]; ap[i + 8] = hi[i]; }
                // B = V: column d = chB + td*16 + nl, k tokens kt*32 + 16*hw .. +15
                const bf16* pb = sVt + (chB + td * 16 + nl) * WS + kt * 32 + hw * 16;
                v16bf bp = *(const v16bf*)pb;
                acc = __builtin_amdgcn_wmma_f32_16x16x32_bf16(
                          false, ap, false, bp, (short)0, acc, false, false);
            }
            // epilogue: depthwise 3x3 LePE conv (per-window zero pad) + bias
            int ch = chB + td * 16 + nl;
            float wloc[9];
            #pragma unroll
            for (int i = 0; i < 9; ++i) wloc[i] = lepe_w[ch * 9 + i];
            float bias = lepe_b[ch];
            #pragma unroll
            for (int v = 0; v < 8; ++v) {
                int s  = tr * 16 + v + 8 * hw;
                int li = s >> 3, lj = s & 7;
                float lep = bias;
                #pragma unroll
                for (int ky = 0; ky < 3; ++ky) {
                    int yi = li + ky - 1;
                    if (yi < 0 || yi >= 8) continue;
                    #pragma unroll
                    for (int kx = 0; kx < 3; ++kx) {
                        int xj = lj + kx - 1;
                        if (xj < 0 || xj >= 8) continue;
                        lep += wloc[ky * 3 + kx] * (float)sVt[ch * WS + yi * 8 + xj];
                    }
                }
                int Lrow = baseSp + li * WG + lj;
                out_x[((size_t)b * LTOK + Lrow) * DIM + ch] = acc[v] + lep;
            }
        }
    }
}

extern "C" void kernel_launch(void* const* d_in, const int* in_sizes, int n_in,
                              void* d_out, int out_size, void* d_ws, size_t ws_size,
                              hipStream_t stream) {
    const float* qkv    = (const float*)d_in[0];   // [3,B,L,C] fp32
    const float* lepe_w = (const float*)d_in[1];   // [C,1,3,3] fp32
    const float* lepe_b = (const float*)d_in[2];   // [C] fp32
    float* out_x = (float*)d_out;                                  // [B,L,C]
    float* out_q = out_x + (size_t)BATCH * LTOK * DIM;             // [nWin,H,WS,HD]
    win_attn_lepe_kernel<<<NWIN, 256, 0, stream>>>(qkv, lepe_w, lepe_b, out_x, out_q);
}
